// QLayer_84980222919435
// MI455X (gfx1250) — matrix operand order repaired and checked by reading, not compiled, not run on hardware
//
#include <hip/hip_runtime.h>
#include <hip/hip_bf16.h>
#include <math.h>

// ---------------------------------------------------------------------------
// QLayer (4-qubit VQC, 2 layers) reformulated for MI455X / gfx1250:
//   out[b, j] = psi(x_b)^T * M_j * psi(x_b)
// psi = 16-element real product state from x[b,0..3]; M_j = Re(U^dag D_j U)
// is a weights-only 16x16 symmetric matrix built once per launch on-device.
// Batched quadratic forms run on v_wmma_f32_16x16x32_f16 (f32 accumulate).
// Kernel is HBM-bound: ~16 MB read (x) + 4 MB write (out) at 23.3 TB/s.
// Hot loop: fully scalarized (no indexed locals), hardware TRANS sin/cos,
// v_permlanex16 cross-half reduce, global prefetch. No LDS.
// ---------------------------------------------------------------------------

typedef _Float16     v16h __attribute__((ext_vector_type(16)));
typedef _Float16     h2   __attribute__((ext_vector_type(2)));
typedef float        v8f  __attribute__((ext_vector_type(8)));
typedef unsigned int v8u  __attribute__((ext_vector_type(8)));

#define TILES_PER_WAVE 4
#define WAVES_PER_BLOCK 8
#define ROWS_PER_BLOCK (WAVES_PER_BLOCK * TILES_PER_WAVE * 16)  // 512

// sin(t/2), cos(t/2) via hardware trans ops: v_sin/cos take turns, so the
// argument is t * (1/2) * (1/2pi) = t * (1/4pi).
#define INV_4PI 0.07957747154594767f

__device__ __forceinline__ float hw_sin_turns(float x) {
#if __has_builtin(__builtin_amdgcn_sinf)
  return __builtin_amdgcn_sinf(x);
#else
  return __sinf(x * 6.283185307179586f);
#endif
}
__device__ __forceinline__ float hw_cos_turns(float x) {
#if __has_builtin(__builtin_amdgcn_cosf)
  return __builtin_amdgcn_cosf(x);
#else
  return __cosf(x * 6.283185307179586f);
#endif
}

// Swap values between the two 16-lane halves of a wave32 (lane ^ 16).
__device__ __forceinline__ float xor16(float v) {
#if __has_builtin(__builtin_amdgcn_permlanex16)
  int i = __builtin_bit_cast(int, v);
  i = __builtin_amdgcn_permlanex16(i, i, 0x76543210, 0xfedcba98, false, false);
  return __builtin_bit_cast(float, i);
#else
  return __shfl_xor(v, 16, 32);
#endif
}

// Pack two f32 into a packed-f16 dword (single v_cvt_pk).
__device__ __forceinline__ unsigned int pack2h(float lo, float hi) {
  h2 pk;
  pk[0] = (_Float16)lo;
  pk[1] = (_Float16)hi;
  return __builtin_bit_cast(unsigned int, pk);
}

// ---------------------------------------------------------------------------
// Prepass: apply the circuit's 32 gates to the identity to get U (16x16
// complex, in LDS, ping-pong buffers), form the four M_j, and emit the exact
// per-lane fp16 WMMA A-operand image into workspace:
//   wsA[j][lane][e] = (lane<16) ? M_j[lane&15][e] : 0
// (contiguous K = 16*half + e; lanes 16..31 are the K=16..31 zero padding.)
// One block, 256 threads; runs once per launch, fully rewrites its ws bytes.
// Uses accurate libm cos/sin (one-time cost, fidelity to the reference).
// ---------------------------------------------------------------------------
__global__ __launch_bounds__(256) void qlayer_build_ws(
    const float* __restrict__ w,   // (2,4,3) row-major
    _Float16* __restrict__ wsA)    // 4*32*16 halves = 4 KB
{
  __shared__ float Ur[2][16][16];
  __shared__ float Ui[2][16][16];
  __shared__ float Msh[4][16][16];

  const int tid = threadIdx.x;          // 0..255
  const int r   = tid >> 4;             // row   0..15
  const int col = tid & 15;             // col   0..15

  Ur[0][r][col] = (r == col) ? 1.f : 0.f;
  Ui[0][r][col] = 0.f;
  __syncthreads();

  int cur = 0;
  for (int l = 0; l < 2; ++l) {
    // ---- CNOT ring: (0,1),(1,2),(2,3),(3,0). Row permutation of U. ----
    for (int c = 0; c < 4; ++c) {
      const int tq   = (c + 1) & 3;
      const int nxt  = cur ^ 1;
      const int rsrc = r ^ ((((r >> (3 - c)) & 1) != 0) ? (1 << (3 - tq)) : 0);
      const float vr = Ur[cur][rsrc][col];
      const float vi = Ui[cur][rsrc][col];
      Ur[nxt][r][col] = vr;
      Ui[nxt][r][col] = vi;
      __syncthreads();
      cur = nxt;
    }
    // ---- RX, RY, RZ per qubit. Row-pair update of U. ----
    for (int q = 0; q < 4; ++q) {
      for (int g = 0; g < 3; ++g) {
        const float th = w[l * 12 + q * 3 + g];
        const float ch = cosf(0.5f * th);
        const float sh = sinf(0.5f * th);
        float m00r, m00i, m01r, m01i, m10r, m10i, m11r, m11i;
        if (g == 0) {        // RX: [[c,-is],[-is,c]]
          m00r = ch; m00i = 0.f; m01r = 0.f; m01i = -sh;
          m10r = 0.f; m10i = -sh; m11r = ch; m11i = 0.f;
        } else if (g == 1) { // RY: [[c,-s],[s,c]]
          m00r = ch; m00i = 0.f; m01r = -sh; m01i = 0.f;
          m10r = sh; m10i = 0.f; m11r = ch; m11i = 0.f;
        } else {             // RZ: diag(e^{-it/2}, e^{+it/2})
          m00r = ch; m00i = -sh; m01r = 0.f; m01i = 0.f;
          m10r = 0.f; m10i = 0.f; m11r = ch; m11i = sh;
        }
        const int bit = 3 - q;
        const int prt = r ^ (1 << bit);
        const int hib = (r >> bit) & 1;
        const float ar = Ur[cur][r][col],   ai = Ui[cur][r][col];
        const float br = Ur[cur][prt][col], bi = Ui[cur][prt][col];
        float xr, xi;
        if (!hib) {  // new[r0] = m00*a + m01*b
          xr = m00r * ar - m00i * ai + m01r * br - m01i * bi;
          xi = m00r * ai + m00i * ar + m01r * bi + m01i * br;
        } else {     // new[r1] = m10*b + m11*a   (b = old row with bit=0)
          xr = m10r * br - m10i * bi + m11r * ar - m11i * ai;
          xi = m10r * bi + m10i * br + m11r * ai + m11i * ar;
        }
        const int nxt = cur ^ 1;
        Ur[nxt][r][col] = xr;
        Ui[nxt][r][col] = xi;
        __syncthreads();
        cur = nxt;
      }
    }
  }

  // ---- M_j[a,b] = sum_k d_j(k) * (Ur[k,a]Ur[k,b] + Ui[k,a]Ui[k,b]) ----
  {
    float acc0 = 0.f, acc1 = 0.f, acc2 = 0.f, acc3 = 0.f;
    #pragma unroll
    for (int k = 0; k < 16; ++k) {
      const float p = Ur[cur][k][r] * Ur[cur][k][col] +
                      Ui[cur][k][r] * Ui[cur][k][col];
      acc0 += ((k >> 3) & 1) ? -p : p;   // qubit 0 = MSB
      acc1 += ((k >> 2) & 1) ? -p : p;
      acc2 += ((k >> 1) & 1) ? -p : p;
      acc3 += ((k >> 0) & 1) ? -p : p;
    }
    Msh[0][r][col] = acc0;
    Msh[1][r][col] = acc1;
    Msh[2][r][col] = acc2;
    Msh[3][r][col] = acc3;
  }
  __syncthreads();

  // ---- Emit per-lane A-operand image (contiguous K = 16*half + e, not the ISA table order) ----
  if (tid < 128) {
    const int j  = tid >> 5;       // matrix 0..3
    const int L  = tid & 31;       // lane
    const int mr = L & 15;         // A row (M)
    const int hv = L >> 4;         // half: elements carry K = 16*hv + e (hv=1 zero)
    _Float16* dst = wsA + (j * 32 + L) * 16;
    #pragma unroll
    for (int e = 0; e < 16; ++e)
      dst[e] = (hv == 0) ? (_Float16)Msh[j][mr][e] : (_Float16)0.f;
  }
}

// ---------------------------------------------------------------------------
// Main kernel: one wave handles TILES_PER_WAVE tiles of 16 batch rows.
//   B operand = Psi^T : lane's own psi row in K=0..15 (lanes 0..15); lanes
//                       16..31 carry the K=16..31 zero padding.
//   D = M_j * Psi^T   : lane l holds D[r + 8*(l>=16), l&15]  ->  the readout
//   o_j = psi^T(M_j psi) finishes with a register-local 8-term dot plus one
//   v_permlanex16.  Fully scalarized: no indexed locals anywhere.
// ---------------------------------------------------------------------------
__global__ __launch_bounds__(256) void qlayer_wmma(
    const float4* __restrict__ x4,   // x viewed as float4; row = 4 float4s
    const v16h*  __restrict__ wsA,   // 4 x 32 lanes x 16 halves
    float2*      __restrict__ out2,  // out viewed as float2; row = 2 float2s
    int batch)
{
  const int lane = threadIdx.x & 31;
  const int wave = threadIdx.x >> 5;
  const bool hiHalf = (lane & 16) != 0;  // K=16..31 half-wave
  const int mrow = lane & 15;            // batch row within tile / B column

  // Constant A operands (M_0..M_3), loaded once per wave.
  const v16h aM0 = wsA[0 * 32 + lane];
  const v16h aM1 = wsA[1 * 32 + lane];
  const v16h aM2 = wsA[2 * 32 + lane];
  const v16h aM3 = wsA[3 * 32 + lane];

  const int blockRow0 = blockIdx.x * ROWS_PER_BLOCK;

  #pragma unroll
  for (int t = 0; t < TILES_PER_WAVE; ++t) {
    const int rowBase = blockRow0 + (wave * TILES_PER_WAVE + t) * 16;
    const int b  = rowBase + mrow;
    const int bc = (b < batch) ? b : (batch - 1);

    // Prefetch next tile's angles (global_prefetch_b8 on gfx1250).
    if (t + 1 < TILES_PER_WAVE) {
      const int nb = (bc + 16 < batch) ? (bc + 16) : bc;
      __builtin_prefetch((const void*)&x4[(size_t)nb * 4], 0, 0);
    }

    // theta_0..3 = x[b, 0..3]  (row stride = 16 floats = 4 float4).
    // sin/cos of theta/2 via single hardware TRANS ops (input in turns).
    const float4 th = x4[(size_t)bc * 4];
    const float a0 = th.x * INV_4PI, a1 = th.y * INV_4PI;
    const float a2 = th.z * INV_4PI, a3 = th.w * INV_4PI;
    const float s0 = hw_sin_turns(a0), c0 = hw_cos_turns(a0);
    const float s1 = hw_sin_turns(a1), c1 = hw_cos_turns(a1);
    const float s2 = hw_sin_turns(a2), c2 = hw_cos_turns(a2);
    const float s3 = hw_sin_turns(a3), c3 = hw_cos_turns(a3);

    // psi[k] = prod_j (bit_j(k) ? sin : cos), qubit 0 = MSB of k.
    // Two-level factoring: psi[4*hi2 + lo2] = w(hi2) * u(lo2).  24 muls.
    const float u00 = c2 * c3, u01 = c2 * s3, u10 = s2 * c3, u11 = s2 * s3;
    const float w00 = c0 * c1, w01 = c0 * s1, w10 = s0 * c1, w11 = s0 * s1;
    const float p00 = w00 * u00, p01 = w00 * u01, p02 = w00 * u10, p03 = w00 * u11;
    const float p04 = w01 * u00, p05 = w01 * u01, p06 = w01 * u10, p07 = w01 * u11;
    const float p08 = w10 * u00, p09 = w10 * u01, p10 = w10 * u10, p11 = w10 * u11;
    const float p12 = w11 * u00, p13 = w11 * u01, p14 = w11 * u10, p15 = w11 * u11;

    // The 8 psi values this lane dots against D (its row half) — one vcc,
    // selected once per tile, reused across all four j.
    const float h0 = hiHalf ? p08 : p00;
    const float h1 = hiHalf ? p09 : p01;
    const float h2v = hiHalf ? p10 : p02;
    const float h3 = hiHalf ? p11 : p03;
    const float h4 = hiHalf ? p12 : p04;
    const float h5 = hiHalf ? p13 : p05;
    const float h6 = hiHalf ? p14 : p06;
    const float h7 = hiHalf ? p15 : p07;

    // B operand as 8 packed dwords (one v_cvt_pk + one cndmask each),
    // assembled with a pure register bit_cast (no memory aggregate).
    v8u bw;
    bw[0] = hiHalf ? 0u : pack2h(p00, p01);
    bw[1] = hiHalf ? 0u : pack2h(p02, p03);
    bw[2] = hiHalf ? 0u : pack2h(p04, p05);
    bw[3] = hiHalf ? 0u : pack2h(p06, p07);
    bw[4] = hiHalf ? 0u : pack2h(p08, p09);
    bw[5] = hiHalf ? 0u : pack2h(p10, p11);
    bw[6] = hiHalf ? 0u : pack2h(p12, p13);
    bw[7] = hiHalf ? 0u : pack2h(p14, p15);
    const v16h bp = __builtin_bit_cast(v16h, bw);
    const v8f cz = {0.f, 0.f, 0.f, 0.f, 0.f, 0.f, 0.f, 0.f};

    // D_j = M_j * Psi^T (f32 accumulate), then register-local readout dot.
    float o0, o1, o2, o3;
    {
      const v8f d = __builtin_amdgcn_wmma_f32_16x16x32_f16(
          false, aM0, false, bp, (short)0, cz, false, false);
      float p = h0 * d[0] + h1 * d[1] + h2v * d[2] + h3 * d[3]
              + h4 * d[4] + h5 * d[5] + h6 * d[6] + h7 * d[7];
      o0 = p + xor16(p);
    }
    {
      const v8f d = __builtin_amdgcn_wmma_f32_16x16x32_f16(
          false, aM1, false, bp, (short)0, cz, false, false);
      float p = h0 * d[0] + h1 * d[1] + h2v * d[2] + h3 * d[3]
              + h4 * d[4] + h5 * d[5] + h6 * d[6] + h7 * d[7];
      o1 = p + xor16(p);
    }
    {
      const v8f d = __builtin_amdgcn_wmma_f32_16x16x32_f16(
          false, aM2, false, bp, (short)0, cz, false, false);
      float p = h0 * d[0] + h1 * d[1] + h2v * d[2] + h3 * d[3]
              + h4 * d[4] + h5 * d[5] + h6 * d[6] + h7 * d[7];
      o2 = p + xor16(p);
    }
    {
      const v8f d = __builtin_amdgcn_wmma_f32_16x16x32_f16(
          false, aM3, false, bp, (short)0, cz, false, false);
      float p = h0 * d[0] + h1 * d[1] + h2v * d[2] + h3 * d[3]
              + h4 * d[4] + h5 * d[5] + h6 * d[6] + h7 * d[7];
      o3 = p + xor16(p);
    }

    // out[b, 0..3]: lanes 0..15 write (o0,o1), lanes 16..31 write (o2,o3)
    // — one single-vcc float2 select, then a fully converged 8B store.
    if (b < batch) {
      float2 v;
      v.x = hiHalf ? o2 : o0;
      v.y = hiHalf ? o3 : o1;
      out2[(size_t)b * 2 + (hiHalf ? 1 : 0)] = v;
    }
  }
}

// ---------------------------------------------------------------------------
extern "C" void kernel_launch(void* const* d_in, const int* in_sizes, int n_in,
                              void* d_out, int out_size, void* d_ws, size_t ws_size,
                              hipStream_t stream) {
  const float* x = (const float*)d_in[0];        // (BATCH, 16) f32
  const float* w = (const float*)d_in[1];        // (2, 4, 3)   f32
  float* out     = (float*)d_out;                // (BATCH, 4)  f32
  _Float16* wsA  = (_Float16*)d_ws;              // 4 KB A-operand images

  const int batch = in_sizes[0] / 16;

  qlayer_build_ws<<<1, 256, 0, stream>>>(w, wsA);

  const int grid = (batch + ROWS_PER_BLOCK - 1) / ROWS_PER_BLOCK;  // 512
  qlayer_wmma<<<grid, 256, 0, stream>>>(
      (const float4*)x, (const v16h*)wsA, (float2*)out, batch);
}
